// PTBModel_23940147708272
// MI455X (gfx1250) — matrix operand order repaired and checked by reading, not compiled, not run on hardware
//
#include <hip/hip_runtime.h>
#include <hip/hip_bf16.h>

// ---------------------------------------------------------------------------
// Types for CDNA5 WMMA (wave32): bf16 A/B fragments (16 x bf16 = 8 VGPRs),
// f32 C/D accumulator (8 x f32 = 8 VGPRs).
// ---------------------------------------------------------------------------
typedef __attribute__((ext_vector_type(16))) __bf16 v16bf;
typedef __attribute__((ext_vector_type(8)))  float  v8f;
typedef __attribute__((ext_vector_type(4)))  float  f32x4;
typedef __attribute__((ext_vector_type(4)))  __bf16 bf16x4;
typedef __attribute__((ext_vector_type(4)))  int    v4i_t;

union Frag16 {
    v16bf    v;
    unsigned u[8];
};

// Model constants
#define HID   1024
#define EMB   1024
#define VOCAB 32000
#define BATCH 32
#define SEQT  128
#define G4    (4 * HID)      // 4096
#define NROW  (BATCH * SEQT) // 4096

// GEMM tiling
#define BM 128
#define BN 128
#define BK 32
#define APAD 8   // pad rows to 80B (16B multiple, conflict-free strides)
#define BPAD 8   // pad rows to 272B

// Async global->LDS path (gfx1250), guarded so we always compile.
#if defined(__gfx1250__) &&                                                   \
    __has_builtin(__builtin_amdgcn_global_load_async_to_lds_b128) &&          \
    __has_builtin(__builtin_amdgcn_s_wait_asynccnt)
#define USE_ASYNC_LDS 1
#else
#define USE_ASYNC_LDS 0
#endif

#if USE_ASYNC_LDS
// Signature (from clang diagnostic): (v4i AS(1)* src, v4i AS(3)* dst,
// imm int offset, imm int cpol).
#define GASYNC_B128(dst, src)                                                 \
    __builtin_amdgcn_global_load_async_to_lds_b128(                           \
        (__attribute__((address_space(1))) v4i_t*)(void*)(src),               \
        (__attribute__((address_space(3))) v4i_t*)(void*)(dst), 0, 0)
#endif

// ---------------------------------------------------------------------------
// Helpers
// ---------------------------------------------------------------------------
__device__ __forceinline__ float fsig(float x) {
    return 1.0f / (1.0f + __expf(-x));
}
__device__ __forceinline__ float ftanh(float x) {
    float ax = fabsf(x);
    float e  = __expf(-2.0f * ax);
    float r  = 1.0f - 2.0f * e / (1.0f + e);
    return copysignf(r, x);
}

// Sense-reversal grid barrier (all blocks must be co-resident; 64 blocks).
__device__ __forceinline__ void grid_barrier(unsigned* counter, unsigned* gen,
                                             unsigned nblk) {
    __threadfence();      // make this thread's global stores agent-visible
    __syncthreads();
    if (threadIdx.x == 0) {
        unsigned g = __hip_atomic_load(gen, __ATOMIC_ACQUIRE,
                                       __HIP_MEMORY_SCOPE_AGENT);
        unsigned arrived = __hip_atomic_fetch_add(counter, 1u, __ATOMIC_ACQ_REL,
                                                  __HIP_MEMORY_SCOPE_AGENT);
        if (arrived == nblk - 1) {
            __hip_atomic_store(counter, 0u, __ATOMIC_RELAXED,
                               __HIP_MEMORY_SCOPE_AGENT);
            __hip_atomic_fetch_add(gen, 1u, __ATOMIC_ACQ_REL,
                                   __HIP_MEMORY_SCOPE_AGENT);
        } else {
            while (__hip_atomic_load(gen, __ATOMIC_ACQUIRE,
                                     __HIP_MEMORY_SCOPE_AGENT) == g) {
                __builtin_amdgcn_s_sleep(2);
            }
        }
    }
    __syncthreads();
}

// ---------------------------------------------------------------------------
// Embedding gather: fp32 rows -> bf16 activation rows. One block per token.
// ---------------------------------------------------------------------------
__global__ __launch_bounds__(256) void embed_bf16(
    const int* __restrict__ seq, const float* __restrict__ emb,
    __bf16* __restrict__ out) {
    int r = blockIdx.x;
    int tok = seq[r];
    const float* src = emb + (size_t)tok * EMB;
    __bf16* dst = out + (size_t)r * EMB;
    for (int c = threadIdx.x; c < EMB; c += 256) dst[c] = (__bf16)src[c];
}

// fp32 -> bf16 bulk conversion, 4-wide.
__global__ __launch_bounds__(256) void cvt_bf16(
    const float* __restrict__ in, __bf16* __restrict__ out, int n4) {
    int i = blockIdx.x * 256 + threadIdx.x;
    int stride = gridDim.x * 256;
    for (; i < n4; i += stride) {
        f32x4 v = ((const f32x4*)in)[i];
        ((bf16x4*)out)[i] = __builtin_convertvector(v, bf16x4);
    }
}

// ---------------------------------------------------------------------------
// bf16 x bf16 -> f32 GEMM with bias: C[M,N] = A[M,K] * B[K,N] + bias[N].
// Block tile 128x128x32, 8 waves, each wave 32x64 (2x4 tiles of 16x16).
// Async path: double-buffered LDS tiles filled by GLOBAL_LOAD_ASYNC_TO_LDS,
// one s_barrier per K-step, paced by s_wait_asynccnt.
// ---------------------------------------------------------------------------
__global__ __launch_bounds__(256) void gemm_bf16f32(
    const __bf16* __restrict__ A, const __bf16* __restrict__ B,
    const float* __restrict__ bias, float* __restrict__ C,
    int M, int N, int K) {
    const int tid   = threadIdx.x;
    const int lane  = tid & 31;
    const int wave  = tid >> 5;
    const int wm    = wave & 3;   // 4 M positions of 32 rows
    const int wn    = wave >> 2;  // 2 N positions of 64 cols
    const int bm0   = blockIdx.y * BM;
    const int bn0   = blockIdx.x * BN;
    const int m16   = lane & 15;
    const int lhalf = lane >> 4;

    v8f acc[2][4];
#pragma unroll
    for (int im = 0; im < 2; ++im)
#pragma unroll
        for (int in = 0; in < 4; ++in) {
            v8f z = {0.f, 0.f, 0.f, 0.f, 0.f, 0.f, 0.f, 0.f};
            acc[im][in] = z;
        }

#if USE_ASYNC_LDS
    __shared__ __bf16 As[2][BM][BK + APAD];
    __shared__ __bf16 Bs[2][BK][BN + BPAD];
#else
    __shared__ __bf16 As[1][BM][BK + APAD];
    __shared__ __bf16 Bs[1][BK][BN + BPAD];
#endif

    // Compute on one resident tile pair: load fragments, 8 WMMAs.
    auto compute_tile = [&](const __bf16 (&AsT)[BM][BK + APAD],
                            const __bf16 (&BsT)[BK][BN + BPAD]) {
        Frag16 a[2], b[4];
#pragma unroll
        for (int im = 0; im < 2; ++im) {
            int row = wm * 32 + im * 16 + m16;
#pragma unroll
            for (int p = 0; p < 8; ++p) {
                int kb = ((p >> 2) << 4) + lhalf * 8 + ((p & 3) << 1);
                a[im].u[p] = *(const unsigned*)&AsT[row][kb];
            }
        }
        {
#pragma unroll
            for (int in = 0; in < 4; ++in) {
                int col = wn * 64 + in * 16 + m16;
#pragma unroll
                for (int s = 0; s < 16; ++s)
                    b[in].v[s] = BsT[((s >> 3) << 4) + lhalf * 8 + (s & 7)][col];
            }
        }
#pragma unroll
        for (int im = 0; im < 2; ++im)
#pragma unroll
            for (int in = 0; in < 4; ++in)
                acc[im][in] = __builtin_amdgcn_wmma_f32_16x16x32_bf16(
                    false, a[im].v, false, b[in].v, (short)0, acc[im][in],
                    false, false);
    };

#if USE_ASYNC_LDS
    // Issue one tile's async loads: A = 512 granules of 16B, B = 512 granules.
    // Each thread issues 2 A-granules + 2 B-granules (b128 each).
    auto issue_tile = [&](int buf, int kk) {
#pragma unroll
        for (int s = 0; s < 2; ++s) {
            int g = tid + s * 256;
            int row = g >> 2, c = (g & 3) * 8;          // A: 4 granules/row
            GASYNC_B128(&As[buf][row][c],
                        A + (size_t)(bm0 + row) * K + kk + c);
        }
#pragma unroll
        for (int s = 0; s < 2; ++s) {
            int g = tid + s * 256;
            int row = g >> 4, c = (g & 15) * 8;         // B: 16 granules/row
            GASYNC_B128(&Bs[buf][row][c],
                        B + (size_t)(kk + row) * N + bn0 + c);
        }
    };

    issue_tile(0, 0);
    const int nt = K / BK;
    for (int i = 0; i < nt; ++i) {
        const int cur = i & 1;
        __builtin_amdgcn_s_wait_asynccnt(0);  // my granules for tile i landed
        __syncthreads();                      // everyone's landed; prev compute done
        if (i + 1 < nt) issue_tile(cur ^ 1, (i + 1) * BK);
        compute_tile(As[cur], Bs[cur]);
    }
#else
    for (int kk = 0; kk < K; kk += BK) {
        {   // A tile 128x32 (16 bf16 per thread, 2x b128)
            int r = tid >> 1, c = (tid & 1) * 16;
            const uint4* src =
                (const uint4*)(A + (size_t)(bm0 + r) * K + kk + c);
            uint4 x0 = src[0], x1 = src[1];
            *(uint4*)&As[0][r][c]     = x0;
            *(uint4*)&As[0][r][c + 8] = x1;
        }
        {   // B tile 32x128
            int r = tid >> 3, c = (tid & 7) * 16;
            const uint4* src =
                (const uint4*)(B + (size_t)(kk + r) * N + bn0 + c);
            uint4 x0 = src[0], x1 = src[1];
            *(uint4*)&Bs[0][r][c]     = x0;
            *(uint4*)&Bs[0][r][c + 8] = x1;
        }
        __syncthreads();
        compute_tile(As[0], Bs[0]);
        __syncthreads();
    }
#endif

    // Epilogue: C/D layout -> M = v + 8*lhalf, N = m16
#pragma unroll
    for (int im = 0; im < 2; ++im) {
        int rowbase = bm0 + wm * 32 + im * 16 + lhalf * 8;
#pragma unroll
        for (int in = 0; in < 4; ++in) {
            int col  = bn0 + wn * 64 + in * 16 + m16;
            float bv = bias[col];
#pragma unroll
            for (int v = 0; v < 8; ++v)
                C[(size_t)(rowbase + v) * N + col] = acc[im][in][v] + bv;
        }
    }
}

// ---------------------------------------------------------------------------
// Persistent LSTM scan: 64 blocks x 256 threads; block owns 16 hidden units.
// Each step: z = xz[:,t,:] + h_{t-1} @ Wh via WMMA, gates, c/h update.
// h double-buffered in global bf16; c resident in LDS; grid barrier per step.
// ---------------------------------------------------------------------------
__global__ __launch_bounds__(256) void lstm_scan(
    const float* __restrict__ xz,   // [B*T, 4H], row = b*T + t
    const __bf16* __restrict__ Wh,  // [H, 4H] bf16
    __bf16* __restrict__ xout,      // [B*T, H] layer output bf16
    __bf16* __restrict__ hbuf,      // [2][B*H] bf16 h state (double buffer)
    unsigned* __restrict__ bar) {
    __shared__ float zs[4][32][16];
    __shared__ float cs[32][16];

    const int tid   = threadIdx.x;
    const int lane  = tid & 31;
    const int wave  = tid >> 5;
    const int g     = wave >> 1;  // gate 0..3 (i,f,g,o)
    const int mh    = wave & 1;   // batch half (rows 0-15 / 16-31)
    const int m16   = lane & 15;
    const int lhalf = lane >> 4;
    const int u0    = blockIdx.x * 16;  // hidden-unit chunk

    for (int i = tid; i < 512; i += 256) cs[i >> 4][i & 15] = 0.0f;
    __syncthreads();

    const int arow = (mh * 16 + m16) * HID;
    const int bcol = g * HID + u0;

    for (int t = 0; t < SEQT; ++t) {
        v8f acc = {0.f, 0.f, 0.f, 0.f, 0.f, 0.f, 0.f, 0.f};
        if (t > 0) {
            const __bf16* hb = hbuf + (size_t)(t & 1) * (BATCH * HID);
#pragma unroll 4
            for (int kk = 0; kk < HID; kk += 32) {
                Frag16 a, b;
#pragma unroll
                for (int p = 0; p < 8; ++p) {
                    int kb = ((p >> 2) << 4) + lhalf * 8 + ((p & 3) << 1);
                    a.u[p] = *(const unsigned*)(hb + arow + kk + kb);
                }
#pragma unroll
                for (int s = 0; s < 16; ++s)
                    b.v[s] = Wh[(size_t)(kk + ((s >> 3) << 4) + lhalf * 8 + (s & 7)) * G4 +
                                bcol + m16];
                acc = __builtin_amdgcn_wmma_f32_16x16x32_bf16(
                    false, a.v, false, b.v, (short)0, acc, false, false);
            }
        }
        // Park z tile in LDS (M = v + 8*lhalf within the 16-row tile)
        {
            int rowb = mh * 16 + lhalf * 8;
#pragma unroll
            for (int v = 0; v < 8; ++v) zs[g][rowb + v][m16] = acc[v];
        }
        __syncthreads();

        // Gates + state update: 512 (batch,unit) cells, 2 per thread
#pragma unroll
        for (int e = 0; e < 2; ++e) {
            int idx = tid * 2 + e;
            int br  = idx >> 4;
            int u   = idx & 15;
            size_t xr = (size_t)(br * SEQT + t) * G4 + u0 + u;
            float zi = zs[0][br][u] + xz[xr];
            float zf = zs[1][br][u] + xz[xr + HID];
            float zg = zs[2][br][u] + xz[xr + 2 * HID];
            float zo = zs[3][br][u] + xz[xr + 3 * HID];
            float ig = fsig(zi), fg = fsig(zf), gg = ftanh(zg), og = fsig(zo);
            float c  = fg * cs[br][u] + ig * gg;
            cs[br][u] = c;
            float h  = og * ftanh(c);
            __bf16 hv = (__bf16)h;
            hbuf[(size_t)((t + 1) & 1) * (BATCH * HID) + br * HID + u0 + u] = hv;
            xout[(size_t)(br * SEQT + t) * HID + u0 + u] = hv;
        }
        grid_barrier(bar, bar + 1, gridDim.x);
    }
}

// ---------------------------------------------------------------------------
// Workspace layout (bytes). Peak ~88.2 MB; dense_w bf16 aliases retired xz.
// ---------------------------------------------------------------------------
#define XZ_OFF  ((size_t)0)                        // 64 MB: xz f32 / dense_w bf16
#define XA_OFF  ((size_t)67108864)                 // 8 MB
#define XB_OFF  (XA_OFF + (size_t)8388608)         // 8 MB
#define W_OFF   (XB_OFF + (size_t)8388608)         // 8 MB (Wx/Wh bf16, reused)
#define H_OFF   (W_OFF + (size_t)8388608)          // 128 KB h double buffer
#define BAR_OFF (H_OFF + (size_t)131072)           // 8 B barrier state

extern "C" void kernel_launch(void* const* d_in, const int* in_sizes, int n_in,
                              void* d_out, int out_size, void* d_ws,
                              size_t ws_size, hipStream_t stream) {
    (void)in_sizes; (void)n_in; (void)out_size; (void)ws_size;
    const int*   seq  = (const int*)d_in[0];
    const float* emb  = (const float*)d_in[1];
    const float* Wx   = (const float*)d_in[2];
    const float* Wh   = (const float*)d_in[3];
    const float* bvec = (const float*)d_in[4];
    const float* dW   = (const float*)d_in[5];
    const float* db   = (const float*)d_in[6];
    float* out = (float*)d_out;
    char*  ws  = (char*)d_ws;

    float*    xz   = (float*)(ws + XZ_OFF);
    __bf16*   dwb  = (__bf16*)(ws + XZ_OFF);   // alias, used after layers
    __bf16*   xA   = (__bf16*)(ws + XA_OFF);
    __bf16*   xB   = (__bf16*)(ws + XB_OFF);
    __bf16*   wbuf = (__bf16*)(ws + W_OFF);
    __bf16*   hbuf = (__bf16*)(ws + H_OFF);
    unsigned* bar  = (unsigned*)(ws + BAR_OFF);

    (void)hipMemsetAsync(bar, 0, 2 * sizeof(unsigned), stream);

    embed_bf16<<<NROW, 256, 0, stream>>>(seq, emb, xA);

    for (int l = 0; l < 2; ++l) {
        cvt_bf16<<<2048, 256, 0, stream>>>(Wx + (size_t)l * EMB * G4, wbuf,
                                           (EMB * G4) / 4);
        gemm_bf16f32<<<dim3(G4 / BN, NROW / BM), 256, 0, stream>>>(
            xA, wbuf, bvec + (size_t)l * G4, xz, NROW, G4, EMB);
        cvt_bf16<<<2048, 256, 0, stream>>>(Wh + (size_t)l * HID * G4, wbuf,
                                           (HID * G4) / 4);
        lstm_scan<<<64, 256, 0, stream>>>(xz, wbuf, xB, hbuf, bar);
        __bf16* tmp = xA; xA = xB; xB = tmp;
    }

    cvt_bf16<<<4096, 256, 0, stream>>>(dW, dwb, (HID * VOCAB) / 4);
    gemm_bf16f32<<<dim3(VOCAB / BN, NROW / BM), 256, 0, stream>>>(
        xA, dwb, db, out, NROW, VOCAB, HID);
}
